// EEGModel_79491254714426
// MI455X (gfx1250) — matrix operand order repaired and checked by reading, not compiled, not run on hardware
//
#include <hip/hip_runtime.h>

// ---------------------------------------------------------------------------
// EEG model for MI455X (gfx1250, wave32, WMMA). Compile-only target.
// All GEMM-shaped stages use v_wmma_f32_16x16x32_f16 (f16 in, f32 accum),
// dual accumulators to break WMMA D->C dependency chains, and
// global_prefetch_b8 (via __builtin_prefetch) on the K-stream.
// ---------------------------------------------------------------------------

typedef __attribute__((ext_vector_type(16))) _Float16 v16h;
typedef __attribute__((ext_vector_type(8)))  float    v8f;

#define B_  16
#define T_  128
#define C_  64
#define F_  128
#define H_  4
#define FO_ 16
#define HFO 64      // H_*FO_
#define HID_ 32
#define NSEQ 1024   // B_*C_
#define TEMP_INV (1.0f/1.5f)

union Frag { v16h v; uint4 q[2]; };

__device__ __forceinline__ void load_a_frag(Frag& a, const _Float16* arow, int k0, int hi)
{
    // A 16x32 f16 fragment: elems 0..7 -> K = hi*8+e ; elems 8..15 -> K = 16+hi*8+e
    a.q[0] = *(const uint4*)(arow + k0 + hi * 8);
    a.q[1] = *(const uint4*)(arow + k0 + 16 + hi * 8);
}
__device__ __forceinline__ void load_b_frag(Frag& b, const _Float16* brow, int k0, int hi)
{
    // B 32x16 f16 fragment: elems 0..7 -> K = hi*8+e ; elems 8..15 -> K = 16+hi*8+e
    b.q[0] = *(const uint4*)(brow + k0 + hi * 8);
    b.q[1] = *(const uint4*)(brow + k0 + 16 + hi * 8);
}

// ---------------------------------------------------------------------------
// Generic NT GEMM:  C[M,N] = A[M,K] * B[N,K]^T  (+bias[n]),  f16 A/B, f32 acc.
// One wave per 16x16 tile, K streamed in steps of 64 (2 independent WMMA
// accumulator chains). M,N mult of 16, K of 32.
// ---------------------------------------------------------------------------
__global__ void eeg_gemm_nt(const _Float16* __restrict__ A,
                            const _Float16* __restrict__ Bm,
                            void* __restrict__ Cout,
                            const float* __restrict__ bias,
                            int M, int N, int K,
                            int lda, int ldb, int ldc,
                            long long sA, long long sB, long long sC,
                            int out_half)
{
    int wave = threadIdx.x >> 5;
    int lane = threadIdx.x & 31;
    int tilesN = N >> 4;
    int tilesM = M >> 4;
    int tile = blockIdx.x * (blockDim.x >> 5) + wave;
    if (tile >= tilesM * tilesN) return;          // wave-uniform: EXEC all-1 at WMMA
    int tm = tile / tilesN, tn = tile % tilesN;
    long long batch = blockIdx.y;

    const _Float16* Ab = A  + batch * sA;
    const _Float16* Bb = Bm + batch * sB;

    int r  = lane & 15;
    int hi = lane >> 4;
    const _Float16* arow = Ab + (size_t)(tm * 16 + r) * lda;
    const _Float16* brow = Bb + (size_t)(tn * 16 + r) * ldb;

    v8f acc0 = {}, acc1 = {};
    int k0 = 0;
    for (; k0 + 64 <= K; k0 += 64) {
        // speculative prefetch ~2 chunks ahead on both K-streams
        __builtin_prefetch(arow + k0 + 128, 0, 1);
        __builtin_prefetch(brow + k0 + 128, 0, 1);
        Frag a0, b0, a1, b1;
        load_a_frag(a0, arow, k0,      hi);
        load_b_frag(b0, brow, k0,      hi);
        load_a_frag(a1, arow, k0 + 32, hi);
        load_b_frag(b1, brow, k0 + 32, hi);
        acc0 = __builtin_amdgcn_wmma_f32_16x16x32_f16(
            false, a0.v, false, b0.v, (short)0, acc0, false, false);
        acc1 = __builtin_amdgcn_wmma_f32_16x16x32_f16(
            false, a1.v, false, b1.v, (short)0, acc1, false, false);
    }
    for (; k0 < K; k0 += 32) {
        Frag a, b;
        load_a_frag(a, arow, k0, hi);
        load_b_frag(b, brow, k0, hi);
        acc0 = __builtin_amdgcn_wmma_f32_16x16x32_f16(
            false, a.v, false, b.v, (short)0, acc0, false, false);
    }
    #pragma unroll
    for (int e = 0; e < 8; ++e) acc0[e] += acc1[e];

    // C layout: elem e -> (m = hi*8+e, n = r)
    int nGlob = tn * 16 + r;
    float bv = bias ? bias[nGlob] : 0.0f;
    if (out_half) {
        _Float16* Ch = (_Float16*)Cout + batch * sC;
        #pragma unroll
        for (int e = 0; e < 8; ++e) {
            int mGlob = tm * 16 + hi * 8 + e;
            Ch[(size_t)mGlob * ldc + nGlob] = (_Float16)(acc0[e] + bv);
        }
    } else {
        float* Cf = (float*)Cout + batch * sC;
        #pragma unroll
        for (int e = 0; e < 8; ++e) {
            int mGlob = tm * 16 + hi * 8 + e;
            Cf[(size_t)mGlob * ldc + nGlob] = acc0[e] + bv;
        }
    }
}

// ---------------------------------------------------------------------------
// Spectral weighting + row normalization. One block per (b,t,c) row of F=128.
// Writes xw16 [b,t,c,f] and Hn16 transposed to [b,c,t,f] for the corr GEMM.
// ---------------------------------------------------------------------------
__global__ void eeg_spec(const float* __restrict__ x, const float* __restrict__ spec_w,
                         _Float16* __restrict__ xw16, _Float16* __restrict__ hn16)
{
    __shared__ float red[F_];
    long long row = blockIdx.x;          // over B*T*C
    int f = threadIdx.x;
    int c = (int)(row % C_);
    long long bt = row / C_;
    int t = (int)(bt % T_);
    int b = (int)(bt / T_);

    float v = x[row * F_ + f] * (1.0f / (1.0f + __expf(-spec_w[c * F_ + f])));
    xw16[row * F_ + f] = (_Float16)v;
    red[f] = v * v;
    __syncthreads();
    for (int s = 64; s > 0; s >>= 1) { if (f < s) red[f] += red[f + s]; __syncthreads(); }
    float inv = 1.0f / (sqrtf(red[0]) + 1e-8f);
    size_t o = (((size_t)b * C_ + c) * T_ + t) * F_ + f;
    hn16[o] = (_Float16)(v * inv);
}

// ---------------------------------------------------------------------------
// Graph: scale/symmetrize corr, top-(k+1) per row, threshold, symmetrize keep,
// row-normalized An and mask bits (incl. diagonal). One block (64 thr) per b.
// ---------------------------------------------------------------------------
__global__ void eeg_graph(const float* __restrict__ Araw, float* __restrict__ An,
                          unsigned long long* __restrict__ maskbits)
{
    __shared__ float sA[C_ * C_];
    __shared__ unsigned long long skeep[C_];
    int b = blockIdx.x, i = threadIdx.x;
    const float* Ab = Araw + b * C_ * C_;
    for (int j = 0; j < C_; ++j)
        sA[i * C_ + j] = 0.5f * (Ab[i * C_ + j] + Ab[j * C_ + i]) * (1.0f / (float)T_);
    __syncthreads();

    unsigned long long sel = 0;
    for (int r = 0; r < 3; ++r) {               // TOPK+1 = 3
        float mv = -1e30f; int mi = 0;
        for (int j = 0; j < C_; ++j)
            if (!((sel >> j) & 1)) { float v = sA[i * C_ + j]; if (v > mv) { mv = v; mi = j; } }
        sel |= 1ull << mi;
    }
    unsigned long long kb = 0;
    for (int j = 0; j < C_; ++j)
        if (((sel >> j) & 1) && sA[i * C_ + j] > 0.3f && j != i) kb |= 1ull << j;
    skeep[i] = kb;
    __syncthreads();
    for (int j = 0; j < C_; ++j) if ((skeep[j] >> i) & 1) kb |= 1ull << j;

    float rs = 0.0f;
    for (int j = 0; j < C_; ++j) if ((kb >> j) & 1) rs += sA[i * C_ + j];
    float inv = 1.0f / (rs + 1e-8f);
    for (int j = 0; j < C_; ++j)
        An[b * C_ * C_ + i * C_ + j] = ((kb >> j) & 1) ? sA[i * C_ + j] * inv : 0.0f;
    maskbits[b * C_ + i] = kb | (1ull << i);
}

// ---------------------------------------------------------------------------
// Edge logits: ei/ej[b,t,c,h] = Wh[b,t,c,h,:] . gat_a[h, :FO] / [FO:]
// ---------------------------------------------------------------------------
__global__ void eeg_edge(const _Float16* __restrict__ wh16, const float* __restrict__ gat_a,
                         float* __restrict__ ei, float* __restrict__ ej)
{
    long long idx = (long long)blockIdx.x * blockDim.x + threadIdx.x;  // B*T*C*H
    if (idx >= (long long)B_ * T_ * C_ * H_) return;
    int h = (int)(idx & 3);
    long long rc = idx >> 2;
    const _Float16* w = wh16 + rc * HFO + h * FO_;
    const float* a1 = gat_a + h * 2 * FO_;
    float s0 = 0.0f, s1 = 0.0f;
    #pragma unroll
    for (int f = 0; f < FO_; ++f) { float wf = (float)w[f]; s0 += wf * a1[f]; s1 += wf * a1[FO_ + f]; }
    ei[idx] = s0; ej[idx] = s1;
}

// ---------------------------------------------------------------------------
// GAT attention: masked softmax over j + aggregation + residual + LN + ReLU.
// One block (256 thr) per (b,t); thread = (h,i).
// ---------------------------------------------------------------------------
__global__ void eeg_att(const _Float16* __restrict__ wh16, const float* __restrict__ res32,
                        const float* __restrict__ ei, const float* __restrict__ ej,
                        const float* __restrict__ An,
                        const unsigned long long* __restrict__ maskbits,
                        const float* __restrict__ gamma, const float* __restrict__ beta,
                        _Float16* __restrict__ g16)
{
    __shared__ _Float16 swh[C_ * HFO];
    __shared__ float sAn[C_ * C_];
    __shared__ float sej[C_ * H_];
    __shared__ unsigned long long smask[C_];
    __shared__ float rsum[256], rsq[256];
    __shared__ float smean[C_], srstd[C_];

    int bt = blockIdx.x;
    int b  = bt >> 7;                 // / T_
    int tid = threadIdx.x;
    long long rowb = (long long)bt * C_;

    for (int p = tid; p < C_ * HFO; p += 256) {
        swh[p] = wh16[rowb * HFO + p];
        sAn[p] = An[(long long)b * C_ * C_ + p];
    }
    sej[tid] = ej[rowb * H_ + tid];
    if (tid < C_) smask[tid] = maskbits[b * C_ + tid];
    __syncthreads();

    int i = tid & 63, h = tid >> 6;
    unsigned long long m = smask[i];
    float eii = ei[(rowb + i) * H_ + h];

    float smax = -1e30f;
    for (int j = 0; j < C_; ++j) {
        if (!((m >> j) & 1)) continue;
        float v = eii + sej[j * H_ + h];
        v = v > 0.0f ? v : 0.2f * v;
        v = (v + sAn[i * C_ + j]) * TEMP_INV;
        smax = fmaxf(smax, v);
    }
    float acc[FO_];
    #pragma unroll
    for (int f = 0; f < FO_; ++f) acc[f] = 0.0f;
    float den = 0.0f;
    for (int j = 0; j < C_; ++j) {
        if (!((m >> j) & 1)) continue;
        float v = eii + sej[j * H_ + h];
        v = v > 0.0f ? v : 0.2f * v;
        v = (v + sAn[i * C_ + j]) * TEMP_INV;
        float w = __expf(v - smax);
        den += w;
        const _Float16* wr = &swh[j * HFO + h * FO_];
        #pragma unroll
        for (int f = 0; f < FO_; ++f) acc[f] += w * (float)wr[f];
    }
    float inv = 1.0f / den;
    float o[FO_], ps = 0.0f, pq = 0.0f;
    const float* rr = res32 + (rowb + i) * HFO + h * FO_;
    #pragma unroll
    for (int f = 0; f < FO_; ++f) {
        float y = acc[f] * inv + rr[f];
        o[f] = y; ps += y; pq += y * y;
    }
    rsum[tid] = ps; rsq[tid] = pq;
    __syncthreads();
    if (h == 0) {
        float S = 0.0f, Q = 0.0f;
        for (int hh = 0; hh < H_; ++hh) { S += rsum[hh * 64 + i]; Q += rsq[hh * 64 + i]; }
        float mean = S * (1.0f / (float)HFO);
        float var  = Q * (1.0f / (float)HFO) - mean * mean;
        smean[i] = mean; srstd[i] = rsqrtf(var + 1e-5f);
    }
    __syncthreads();
    float mean = smean[i], rstd = srstd[i];
    _Float16* gout = g16 + (rowb + i) * HFO + h * FO_;
    #pragma unroll
    for (int f = 0; f < FO_; ++f) {
        float y = (o[f] - mean) * rstd * gamma[h * FO_ + f] + beta[h * FO_ + f];
        gout[f] = (_Float16)(y > 0.0f ? y : 0.0f);
    }
}

// ---------------------------------------------------------------------------
// LSTM recurrence: one sequence per wave, 32 lanes <-> 32 hidden units.
// Gate preacts zin (x@Wih^T + biases) precomputed by WMMA GEMM.
// WhhT staged k-major in LDS (conflict-free); h broadcast via wave32 shuffles.
// ---------------------------------------------------------------------------
__global__ void eeg_lstm(const _Float16* __restrict__ zin, const float* __restrict__ Whh,
                         _Float16* __restrict__ hout, int dir)
{
    __shared__ float sW[HID_ * 4 * HID_];        // [k][j], j over 4*HID
    int tid = threadIdx.x;
    for (int p = tid; p < HID_ * 4 * HID_; p += 256) {
        int k = p >> 7, j = p & 127;
        sW[p] = Whh[j * HID_ + k];
    }
    __syncthreads();

    int wave = tid >> 5, u = tid & 31;
    int n = blockIdx.x * 8 + wave;
    int off = dir ? HID_ : 0;
    float h = 0.0f, c = 0.0f;
    for (int s = 0; s < T_; ++s) {
        int t = dir ? (T_ - 1 - s) : s;
        long long row = (long long)n * T_ + t;
        const _Float16* z = zin + row * (4 * HID_);
        float zi = (float)z[u], zf = (float)z[HID_ + u];
        float zg = (float)z[2 * HID_ + u], zo = (float)z[3 * HID_ + u];
        for (int k = 0; k < HID_; ++k) {
            float hk = __shfl(h, k, 32);
            const float* wk = &sW[k * 4 * HID_];
            zi += hk * wk[u];            zf += hk * wk[HID_ + u];
            zg += hk * wk[2 * HID_ + u]; zo += hk * wk[3 * HID_ + u];
        }
        float ig = 1.0f / (1.0f + __expf(-zi));
        float fg = 1.0f / (1.0f + __expf(-zf));
        float gg = tanhf(zg);
        float og = 1.0f / (1.0f + __expf(-zo));
        c = fg * c + ig * gg;
        h = og * tanhf(c);
        hout[row * (2 * HID_) + off + u] = (_Float16)h;
    }
}

// ---------------------------------------------------------------------------
// Temporal attention score: sc[r] = ta_v . tanh(u[r,:])  (u from WMMA GEMM).
// One wave per row.
// ---------------------------------------------------------------------------
__global__ void eeg_sc(const float* __restrict__ u32, const float* __restrict__ ta_v,
                       float* __restrict__ sc)
{
    int wave = threadIdx.x >> 5, lane = threadIdx.x & 31;
    long long r = (long long)blockIdx.x * 8 + wave;
    const float* ur = u32 + r * (2 * HID_);
    float s = tanhf(ur[lane]) * ta_v[lane] + tanhf(ur[32 + lane]) * ta_v[32 + lane];
    for (int o = 16; o > 0; o >>= 1) s += __shfl_xor(s, o, 32);
    if (lane == 0) sc[r] = s;
}

// ---------------------------------------------------------------------------
// Temporal softmax pooling per sequence: hp[n,f] = sum_t h[n,t,f]*al[n,t].
// ---------------------------------------------------------------------------
__global__ void eeg_pool(const float* __restrict__ sc, const _Float16* __restrict__ h16,
                         float* __restrict__ hp)
{
    __shared__ float sal[T_], red[T_];
    int n = blockIdx.x, tid = threadIdx.x;
    float s = sc[(long long)n * T_ + tid] * TEMP_INV;
    red[tid] = s; __syncthreads();
    for (int k = 64; k > 0; k >>= 1) { if (tid < k) red[tid] = fmaxf(red[tid], red[tid + k]); __syncthreads(); }
    float mx = red[0]; __syncthreads();
    float e = __expf(s - mx);
    red[tid] = e; __syncthreads();
    for (int k = 64; k > 0; k >>= 1) { if (tid < k) red[tid] += red[tid + k]; __syncthreads(); }
    sal[tid] = e / red[0];
    __syncthreads();
    if (tid < 2 * HID_) {
        float a = 0.0f;
        const _Float16* hb = h16 + (long long)n * T_ * (2 * HID_) + tid;
        for (int t = 0; t < T_; ++t) a += (float)hb[t * (2 * HID_)] * sal[t];
        hp[n * (2 * HID_) + tid] = a;
    }
}

// ---------------------------------------------------------------------------
// Head: channel-attention pooling + LN + fc1(relu) + fc2. One block (64) per b.
// ---------------------------------------------------------------------------
__global__ void eeg_head(const float* __restrict__ hp,
                         const float* __restrict__ cp_W, const float* __restrict__ cp_b,
                         const float* __restrict__ norm_g, const float* __restrict__ norm_b,
                         const float* __restrict__ fc1_W, const float* __restrict__ fc1_b,
                         const float* __restrict__ fc2_W, const float* __restrict__ fc2_b,
                         float* __restrict__ out)
{
    __shared__ float sw[C_], red[C_], sp[C_], sh1[32];
    int b = blockIdx.x, c = threadIdx.x;
    const float* xc = hp + (b * C_ + c) * (2 * HID_);
    float s = cp_b[0];
    for (int f = 0; f < 2 * HID_; ++f) s += xc[f] * cp_W[f];
    red[c] = s; __syncthreads();
    for (int k = 32; k > 0; k >>= 1) { if (c < k) red[c] = fmaxf(red[c], red[c + k]); __syncthreads(); }
    float mx = red[0]; __syncthreads();
    float e = __expf(s - mx);
    red[c] = e; __syncthreads();
    for (int k = 32; k > 0; k >>= 1) { if (c < k) red[c] += red[c + k]; __syncthreads(); }
    sw[c] = e / red[0];
    __syncthreads();

    float pf = 0.0f;                               // pooled[feature c]
    for (int cc = 0; cc < C_; ++cc) pf += hp[(b * C_ + cc) * (2 * HID_) + c] * sw[cc];

    red[c] = pf; __syncthreads();
    for (int k = 32; k > 0; k >>= 1) { if (c < k) red[c] += red[c + k]; __syncthreads(); }
    float mean = red[0] * (1.0f / 64.0f); __syncthreads();
    float d = pf - mean;
    red[c] = d * d; __syncthreads();
    for (int k = 32; k > 0; k >>= 1) { if (c < k) red[c] += red[c + k]; __syncthreads(); }
    float rstd = rsqrtf(red[0] * (1.0f / 64.0f) + 1e-5f); __syncthreads();
    sp[c] = d * rstd * norm_g[c] + norm_b[c];
    __syncthreads();
    if (c < 32) {
        float a = fc1_b[c];
        for (int f = 0; f < 64; ++f) a += sp[f] * fc1_W[c * 64 + f];
        sh1[c] = a > 0.0f ? a : 0.0f;
    }
    __syncthreads();
    if (c < 3) {
        float a = fc2_b[c];
        for (int j = 0; j < 32; ++j) a += sh1[j] * fc2_W[c * 32 + j];
        out[b * 3 + c] = a;
    }
}

// ---------------------------------------------------------------------------
// Small utilities
// ---------------------------------------------------------------------------
__global__ void eeg_cvt(const float* __restrict__ s, _Float16* __restrict__ d, int n)
{
    int i = blockIdx.x * blockDim.x + threadIdx.x;
    if (i < n) d[i] = (_Float16)s[i];
}
__global__ void eeg_add2(const float* __restrict__ a, const float* __restrict__ b,
                         float* __restrict__ d, int n)
{
    int i = blockIdx.x * blockDim.x + threadIdx.x;
    if (i < n) d[i] = a[i] + b[i];
}

// ---------------------------------------------------------------------------
// Workspace layout (bytes). Region A is phase-aliased.
// ---------------------------------------------------------------------------
static constexpr long long S1 = (long long)B_ * T_ * C_ * F_;   // 16,777,216
static constexpr long long S2 = (long long)B_ * T_ * C_ * HFO;  //  8,388,608
static constexpr size_t O_XW16 = 0;                      // phase1
static constexpr size_t O_HN16 = O_XW16 + (size_t)S1 * 2;
static constexpr size_t O_RES  = O_HN16 + (size_t)S1 * 2;
static constexpr size_t REGA_END = O_RES + (size_t)S2 * 4;   // 100,663,296
static constexpr size_t O_ZINF = 0;                      // phase2 (f16, S1 elems)
static constexpr size_t O_ZINB = (size_t)S1 * 2;
static constexpr size_t O_U32  = 0;                      // phase3 (f32, S2 elems)
static constexpr size_t O_WH16 = REGA_END;               // also reused as h16
static constexpr size_t O_G16  = O_WH16 + (size_t)S2 * 2;
static constexpr size_t O_EI   = O_G16 + (size_t)S2 * 2;
static constexpr size_t O_EJ   = O_EI + (size_t)B_ * T_ * C_ * H_ * 4;
static constexpr size_t O_ARAW = O_EJ + (size_t)B_ * T_ * C_ * H_ * 4;
static constexpr size_t O_AN   = O_ARAW + (size_t)B_ * C_ * C_ * 4;
static constexpr size_t O_MASK = O_AN + (size_t)B_ * C_ * C_ * 4;
static constexpr size_t O_SC   = O_MASK + (size_t)B_ * C_ * 8;
static constexpr size_t O_HP   = O_SC + (size_t)NSEQ * T_ * 4;
static constexpr size_t O_BZF  = O_HP + (size_t)NSEQ * 2 * HID_ * 4;
static constexpr size_t O_BZB  = O_BZF + 512;
static constexpr size_t O_GATW16 = O_BZB + 512;
static constexpr size_t O_RESW16 = O_GATW16 + 16384;
static constexpr size_t O_WIHF16 = O_RESW16 + 16384;
static constexpr size_t O_WIHB16 = O_WIHF16 + 16384;
static constexpr size_t O_TAW16  = O_WIHB16 + 16384;

static void gemm_nt(const void* A, const void* Bm, void* C, const float* bias,
                    int M, int N, int K, int lda, int ldb, int ldc,
                    long long sA, long long sB, long long sC, int batch, int out_half,
                    hipStream_t st)
{
    int tiles = (M / 16) * (N / 16);
    dim3 grid((tiles + 7) / 8, batch);
    eeg_gemm_nt<<<grid, 256, 0, st>>>((const _Float16*)A, (const _Float16*)Bm, C, bias,
                                      M, N, K, lda, ldb, ldc, sA, sB, sC, out_half);
}

extern "C" void kernel_launch(void* const* d_in, const int* in_sizes, int n_in,
                              void* d_out, int out_size, void* d_ws, size_t ws_size,
                              hipStream_t stream)
{
    (void)in_sizes; (void)n_in; (void)out_size; (void)ws_size;

    const float* x      = (const float*)d_in[0];
    const float* spec_w = (const float*)d_in[1];
    const float* gat_W  = (const float*)d_in[2];
    const float* gat_a  = (const float*)d_in[3];
    const float* gat_ng = (const float*)d_in[4];
    const float* gat_nb = (const float*)d_in[5];
    const float* gat_rW = (const float*)d_in[6];
    const float* gat_rb = (const float*)d_in[7];
    const float* lWihf  = (const float*)d_in[8];
    const float* lWhhf  = (const float*)d_in[9];
    const float* lbihf  = (const float*)d_in[10];
    const float* lbhhf  = (const float*)d_in[11];
    const float* lWihb  = (const float*)d_in[12];
    const float* lWhhb  = (const float*)d_in[13];
    const float* lbihb  = (const float*)d_in[14];
    const float* lbhhb  = (const float*)d_in[15];
    const float* ta_W   = (const float*)d_in[16];
    const float* ta_b   = (const float*)d_in[17];
    const float* ta_v   = (const float*)d_in[18];
    const float* cp_W   = (const float*)d_in[19];
    const float* cp_b   = (const float*)d_in[20];
    const float* norm_g = (const float*)d_in[21];
    const float* norm_b = (const float*)d_in[22];
    const float* fc1_W  = (const float*)d_in[23];
    const float* fc1_b  = (const float*)d_in[24];
    const float* fc2_W  = (const float*)d_in[25];
    const float* fc2_b  = (const float*)d_in[26];

    char* ws = (char*)d_ws;
    _Float16* xw16  = (_Float16*)(ws + O_XW16);
    _Float16* hn16  = (_Float16*)(ws + O_HN16);
    float*    res32 = (float*)   (ws + O_RES);
    _Float16* zinf  = (_Float16*)(ws + O_ZINF);
    _Float16* zinb  = (_Float16*)(ws + O_ZINB);
    float*    u32   = (float*)   (ws + O_U32);
    _Float16* wh16  = (_Float16*)(ws + O_WH16);
    _Float16* h16   = (_Float16*)(ws + O_WH16);   // aliased after wh16 dies
    _Float16* g16   = (_Float16*)(ws + O_G16);
    float*    ei    = (float*)   (ws + O_EI);
    float*    ej    = (float*)   (ws + O_EJ);
    float*    Araw  = (float*)   (ws + O_ARAW);
    float*    An    = (float*)   (ws + O_AN);
    unsigned long long* maskb = (unsigned long long*)(ws + O_MASK);
    float*    sc    = (float*)   (ws + O_SC);
    float*    hp    = (float*)   (ws + O_HP);
    float*    bzf   = (float*)   (ws + O_BZF);
    float*    bzb   = (float*)   (ws + O_BZB);
    _Float16* gatW16 = (_Float16*)(ws + O_GATW16);
    _Float16* resW16 = (_Float16*)(ws + O_RESW16);
    _Float16* wihf16 = (_Float16*)(ws + O_WIHF16);
    _Float16* wihb16 = (_Float16*)(ws + O_WIHB16);
    _Float16* taW16  = (_Float16*)(ws + O_TAW16);

    // --- weight conversions + fused biases ---
    eeg_cvt<<<32, 256, 0, stream>>>(gat_W,  gatW16, HFO * F_);
    eeg_cvt<<<32, 256, 0, stream>>>(gat_rW, resW16, HFO * F_);
    eeg_cvt<<<32, 256, 0, stream>>>(lWihf,  wihf16, 4 * HID_ * HFO);
    eeg_cvt<<<32, 256, 0, stream>>>(lWihb,  wihb16, 4 * HID_ * HFO);
    eeg_cvt<<<16, 256, 0, stream>>>(ta_W,   taW16,  2 * HID_ * 2 * HID_);
    eeg_add2<<<1, 128, 0, stream>>>(lbihf, lbhhf, bzf, 4 * HID_);
    eeg_add2<<<1, 128, 0, stream>>>(lbihb, lbhhb, bzb, 4 * HID_);

    // --- spectral weighting + Hn (transposed to [b,c,t,f]) ---
    eeg_spec<<<B_ * T_ * C_, F_, 0, stream>>>(x, spec_w, xw16, hn16);

    // --- correlation GEMM: per b, 64x64 = Hn(64x16384) . Hn^T  (WMMA) ---
    gemm_nt(hn16, hn16, Araw, nullptr, C_, C_, T_ * F_,
            T_ * F_, T_ * F_, C_,
            (long long)C_ * T_ * F_, (long long)C_ * T_ * F_, (long long)C_ * C_,
            B_, 0, stream);

    // --- graph sparsification ---
    eeg_graph<<<B_, C_, 0, stream>>>(Araw, An, maskb);

    // --- GAT projections (WMMA): Wh (f16 out) and residual (f32 out + bias) ---
    gemm_nt(xw16, gatW16, wh16, nullptr, B_ * T_ * C_, HFO, F_,
            F_, F_, HFO, 0, 0, 0, 1, 1, stream);
    gemm_nt(xw16, resW16, res32, gat_rb, B_ * T_ * C_, HFO, F_,
            F_, F_, HFO, 0, 0, 0, 1, 0, stream);

    // --- edge logits, attention + aggregation + residual + LN + ReLU ---
    eeg_edge<<<(B_ * T_ * C_ * H_ + 255) / 256, 256, 0, stream>>>(wh16, gat_a, ei, ej);
    eeg_att<<<B_ * T_, 256, 0, stream>>>(wh16, res32, ei, ej, An, maskb,
                                         gat_ng, gat_nb, g16);

    // --- LSTM input projections (WMMA), hoisted out of time loop ---
    gemm_nt(g16, wihf16, zinf, bzf, NSEQ * T_, 4 * HID_, HFO,
            HFO, HFO, 4 * HID_, 0, 0, 0, 1, 1, stream);
    gemm_nt(g16, wihb16, zinb, bzb, NSEQ * T_, 4 * HID_, HFO,
            HFO, HFO, 4 * HID_, 0, 0, 0, 1, 1, stream);

    // --- recurrences: one sequence per wave (wave32), fwd + bwd ---
    eeg_lstm<<<NSEQ / 8, 256, 0, stream>>>(zinf, lWhhf, h16, 0);
    eeg_lstm<<<NSEQ / 8, 256, 0, stream>>>(zinb, lWhhb, h16, 1);

    // --- temporal attention projection (WMMA) + score + pooling ---
    gemm_nt(h16, taW16, u32, ta_b, NSEQ * T_, 2 * HID_, 2 * HID_,
            2 * HID_, 2 * HID_, 2 * HID_, 0, 0, 0, 1, 0, stream);
    eeg_sc<<<(NSEQ * T_) / 8, 256, 0, stream>>>(u32, ta_v, sc);
    eeg_pool<<<NSEQ, T_, 0, stream>>>(sc, h16, hp);

    // --- head ---
    eeg_head<<<B_, C_, 0, stream>>>(hp, cp_W, cp_b, norm_g, norm_b,
                                    fc1_W, fc1_b, fc2_W, fc2_b, (float*)d_out);
}
